// GCN_5454608466410
// MI455X (gfx1250) — matrix operand order repaired and checked by reading, not compiled, not run on hardware
//
#include <hip/hip_runtime.h>
#include <hip/hip_bf16.h>

#define NN 50000
#define NE 800000
#define DH 64
#define BN_EPS 1e-5f

typedef __attribute__((ext_vector_type(16))) __bf16 bf16x16;
typedef __attribute__((ext_vector_type(8)))  float  floatx8;

__device__ __forceinline__ __bf16 f2bf(float f) {
  union { float f; unsigned u; } v; v.f = f;
  unsigned r = v.u + 0x7FFFu + ((v.u >> 16) & 1u);   // round-to-nearest-even
  unsigned short h = (unsigned short)(r >> 16);
  __bf16 b;
  __builtin_memcpy(&b, &h, sizeof(b));
  return b;
}

// ---------------------------------------------------------------------------
// H[N,64] = X[N,K] @ W[K,64], via v_wmma_f32_16x16x32_bf16 (f32 accumulate).
// Block = 256 threads = 8 wave32s; each wave computes a 16x64 output strip.
// W is staged in LDS transposed ([col][k], bf16) so each lane's B fragment is
// a contiguous 32-byte LDS read.
// ---------------------------------------------------------------------------
template <int K>
__global__ __launch_bounds__(256) void gcn_gemm_wmma(const float* __restrict__ X,
                                                     const float* __restrict__ W,
                                                     float* __restrict__ H, int n) {
  __shared__ __bf16 WT[64 * K];  // transposed: WT[c*K + k]
  for (int idx = threadIdx.x; idx < K * 64; idx += 256) {
    int k = idx >> 6, c = idx & 63;
    WT[c * K + k] = f2bf(W[idx]);
  }
  __syncthreads();

  const int wave = threadIdx.x >> 5;
  const int lane = threadIdx.x & 31;
  const int r0 = blockIdx.x * 128 + wave * 16;
  if (r0 >= n) return;                 // wave-uniform: EXEC stays all-ones

  const int lhi = lane >> 4;           // 0: lanes 0-15, 1: lanes 16-31
  const int llo = lane & 15;

  floatx8 acc0 = {}, acc1 = {}, acc2 = {}, acc3 = {};

  // A-matrix 16x32 bf16 layout: lane holds row M=lane%16; lane half h holds
  // K = 16h..16h+15 of each 32-K step (contiguous order used by this code, as for B).
  const float* xrow = X + (size_t)(r0 + llo) * K;

  for (int k0 = 0; k0 < K; k0 += 32) {
    if (k0 + 32 < K) __builtin_prefetch(xrow + k0 + 32, 0, 1);  // global_prefetch_b8

    bf16x16 afrag;
    const float* ap = xrow + k0 + lhi * 16;
#pragma unroll
    for (int j = 0; j < 8; ++j) {
      afrag[j]     = f2bf(ap[j]);        // K = k0 + lhi*16 + j
      afrag[8 + j] = f2bf(ap[8 + j]);    // K = k0 + lhi*16 + 8 + j
    }

    // B-matrix 32x16 bf16: lane holds col N=lane%16, K=(lane/16)*16 .. +15
    const int kk = k0 + lhi * 16;
    bf16x16 b0 = *reinterpret_cast<const bf16x16*>(&WT[(0 * 16 + llo) * K + kk]);
    acc0 = __builtin_amdgcn_wmma_f32_16x16x32_bf16(false, afrag, false, b0,
                                                   (short)0, acc0, false, false);
    bf16x16 b1 = *reinterpret_cast<const bf16x16*>(&WT[(1 * 16 + llo) * K + kk]);
    acc1 = __builtin_amdgcn_wmma_f32_16x16x32_bf16(false, afrag, false, b1,
                                                   (short)0, acc1, false, false);
    bf16x16 b2 = *reinterpret_cast<const bf16x16*>(&WT[(2 * 16 + llo) * K + kk]);
    acc2 = __builtin_amdgcn_wmma_f32_16x16x32_bf16(false, afrag, false, b2,
                                                   (short)0, acc2, false, false);
    bf16x16 b3 = *reinterpret_cast<const bf16x16*>(&WT[(3 * 16 + llo) * K + kk]);
    acc3 = __builtin_amdgcn_wmma_f32_16x16x32_bf16(false, afrag, false, b3,
                                                   (short)0, acc3, false, false);
  }

  // C/D layout: VGPR i -> M = i + 8*(lane/16); N = lane%16
#pragma unroll
  for (int i = 0; i < 8; ++i) {
    float* hr = H + (size_t)(r0 + lhi * 8 + i) * DH + llo;
    hr[0]  = acc0[i];
    hr[16] = acc1[i];
    hr[32] = acc2[i];
    hr[48] = acc3[i];
  }
}

// ---------------------------------------------------------------------------
// Graph plumbing kernels
// ---------------------------------------------------------------------------
__global__ void deg_count(const int* __restrict__ dst, float* __restrict__ deg, int e) {
  int t = blockIdx.x * blockDim.x + threadIdx.x;
  if (t < e) unsafeAtomicAdd(&deg[dst[t]], 1.0f);
}

__global__ void deg_finish(float* __restrict__ dinv, int n) {
  int t = blockIdx.x * blockDim.x + threadIdx.x;
  if (t < n) dinv[t] = rsqrtf(dinv[t] + 1.0f);   // +1 self-loop
}

// agg = h * dinv^2 (self-loop term) + bias
__global__ void agg_init(const float* __restrict__ h, const float* __restrict__ dinv,
                         const float* __restrict__ bias, float* __restrict__ agg, int total) {
  int t = blockIdx.x * blockDim.x + threadIdx.x;
  if (t < total) {
    int i = t >> 6, f = t & 63;
    float di = dinv[i];
    agg[t] = h[t] * di * di + bias[f];
  }
}

// thread per (edge, feature): coalesced gather + coalesced f32 atomics into L2
__global__ void edge_scatter(const float* __restrict__ h, const float* __restrict__ dinv,
                             const int* __restrict__ src, const int* __restrict__ dst,
                             float* __restrict__ agg, int total) {
  int t = blockIdx.x * blockDim.x + threadIdx.x;
  if (t < total) {
    int e = t >> 6, f = t & 63;
    int s = src[e], d = dst[e];
    float w = dinv[s] * dinv[d];
    unsafeAtomicAdd(&agg[(size_t)d * DH + f], h[(size_t)s * DH + f] * w);
  }
}

// one block per feature column: biased mean/var over N rows
__global__ __launch_bounds__(256) void bn_stats(const float* __restrict__ a,
                                                float* __restrict__ mu,
                                                float* __restrict__ rsig, int n) {
  __shared__ float ssum[256], ssq[256];
  const int c = blockIdx.x;
  float sum = 0.f, sq = 0.f;
  for (int r = threadIdx.x; r < n; r += 256) {
    float v = a[(size_t)r * DH + c];
    sum += v; sq += v * v;
  }
  ssum[threadIdx.x] = sum; ssq[threadIdx.x] = sq;
  __syncthreads();
  for (int s = 128; s > 0; s >>= 1) {
    if (threadIdx.x < s) {
      ssum[threadIdx.x] += ssum[threadIdx.x + s];
      ssq[threadIdx.x]  += ssq[threadIdx.x + s];
    }
    __syncthreads();
  }
  if (threadIdx.x == 0) {
    float m = ssum[0] / (float)n;
    float var = ssq[0] / (float)n - m * m;
    mu[c] = m;
    rsig[c] = rsqrtf(var + BN_EPS);
  }
}

__global__ void bn_relu(const float* __restrict__ a, const float* __restrict__ mu,
                        const float* __restrict__ rsig, const float* __restrict__ g,
                        const float* __restrict__ b, float* __restrict__ xo, int total) {
  int t = blockIdx.x * blockDim.x + threadIdx.x;
  if (t < total) {
    int f = t & 63;
    float y = (a[t] - mu[f]) * rsig[f] * g[f] + b[f];
    xo[t] = fmaxf(y, 0.f);
  }
}

__global__ void out_proj(const float* __restrict__ x, const float* __restrict__ wout,
                         const float* __restrict__ bout, float* __restrict__ out, int n) {
  int t = blockIdx.x * blockDim.x + threadIdx.x;
  if (t < n) {
    const float* xr = x + (size_t)t * DH;
    float s = 0.f;
#pragma unroll
    for (int f = 0; f < DH; ++f) s += xr[f] * wout[f];
    out[t] = s + bout[0];
  }
}

// ---------------------------------------------------------------------------
extern "C" void kernel_launch(void* const* d_in, const int* in_sizes, int n_in,
                              void* d_out, int out_size, void* d_ws, size_t ws_size,
                              hipStream_t stream) {
  const float* x    = (const float*)d_in[0];
  const int*   ei   = (const int*)d_in[1];
  const float* W0   = (const float*)d_in[2];
  const float* b0   = (const float*)d_in[3];
  const float* W1   = (const float*)d_in[4];
  const float* b1   = (const float*)d_in[5];
  const float* W2   = (const float*)d_in[6];
  const float* b2   = (const float*)d_in[7];
  const float* gam  = (const float*)d_in[8];
  const float* bet  = (const float*)d_in[9];
  const float* Wout = (const float*)d_in[10];
  const float* bout = (const float*)d_in[11];
  float* out = (float*)d_out;

  const int* src = ei;
  const int* dst = ei + NE;

  // workspace layout (floats)
  float* ws   = (float*)d_ws;
  float* dinv = ws;                        // NN
  float* h    = dinv + NN;                 // NN*64
  float* agg  = h + (size_t)NN * DH;       // NN*64
  float* xbuf = agg + (size_t)NN * DH;     // NN*64
  float* mu   = xbuf + (size_t)NN * DH;    // 64
  float* rsig = mu + DH;                   // 64

  const int totalNF = NN * DH;             // 3,200,000
  const int totalEF = NE * DH;             // 51,200,000
  const int gemmBlocks = (NN + 127) / 128; // 391

  // degree -> d^{-1/2}
  hipMemsetAsync(dinv, 0, NN * sizeof(float), stream);
  deg_count<<<(NE + 255) / 256, 256, 0, stream>>>(dst, dinv, NE);
  deg_finish<<<(NN + 255) / 256, 256, 0, stream>>>(dinv, NN);

  const float* Ws[3]  = {W0, W1, W2};
  const float* bs[3]  = {b0, b1, b2};
  const float* xin = x;
  for (int layer = 0; layer < 3; ++layer) {
    if (layer == 0)
      gcn_gemm_wmma<128><<<gemmBlocks, 256, 0, stream>>>(xin, Ws[layer], h, NN);
    else
      gcn_gemm_wmma<64><<<gemmBlocks, 256, 0, stream>>>(xin, Ws[layer], h, NN);

    agg_init<<<(totalNF + 255) / 256, 256, 0, stream>>>(h, dinv, bs[layer], agg, totalNF);
    edge_scatter<<<(totalEF + 255) / 256, 256, 0, stream>>>(h, dinv, src, dst, agg, totalEF);
    bn_stats<<<DH, 256, 0, stream>>>(agg, mu, rsig, NN);
    bn_relu<<<(totalNF + 255) / 256, 256, 0, stream>>>(agg, mu, rsig,
                                                       gam + layer * DH, bet + layer * DH,
                                                       xbuf, totalNF);
    xin = xbuf;
  }

  out_proj<<<(NN + 255) / 256, 256, 0, stream>>>(xbuf, Wout, bout, out, NN);
}
